// CGCLR_72370198937695
// MI455X (gfx1250) — matrix operand order repaired and checked by reading, not compiled, not run on hardware
//
#include <hip/hip_runtime.h>
#include <hip/hip_bf16.h>
#include <stdint.h>

// ---------------------------------------------------------------------------
// Types for CDNA5 WMMA (wave32): v_wmma_f32_16x16x32_bf16
// ---------------------------------------------------------------------------
typedef __attribute__((ext_vector_type(16))) __bf16 v16bf;
typedef __attribute__((ext_vector_type(8)))  float  v8f;

union Frag16 { v16bf v; uint4 q[2]; };

// Async global->LDS path (CDNA5): guarded so the file compiles on both the
// host pass and any toolchain lacking the builtin.
#if defined(__HIP_DEVICE_COMPILE__) && \
    __has_builtin(__builtin_amdgcn_global_load_async_to_lds_b128)
#define HAVE_ASYNC_LDS 1
#else
#define HAVE_ASYNC_LDS 0
#endif

#if HAVE_ASYNC_LDS
// Exact parameter types per clang diagnostic:
//   param0: __attribute__((vector_size(16))) int __device__ *   (AS1)
//   param1: same pointee in __shared__ (AS3)
typedef int v4i_g __attribute__((vector_size(16)));
typedef __attribute__((address_space(1))) v4i_g* async_gptr;
typedef __attribute__((address_space(3))) v4i_g* async_lptr;
#endif

__device__ __forceinline__ void async_b128(const unsigned short* g,
                                           unsigned short* l) {
#if HAVE_ASYNC_LDS
  // global_load_async_to_lds_b128: per-lane 16B, tracked by ASYNCcnt.
  __builtin_amdgcn_global_load_async_to_lds_b128(
      (async_gptr)(uintptr_t)g, (async_lptr)(uintptr_t)l, 0, 0);
#else
  *(uint4*)l = *(const uint4*)g;
#endif
}

template <int N>
__device__ __forceinline__ void wait_async() {
#if HAVE_ASYNC_LDS
#if __has_builtin(__builtin_amdgcn_s_wait_asynccnt)
  __builtin_amdgcn_s_wait_asynccnt(N);
#else
  asm volatile("s_wait_asynccnt %0" ::"n"(N) : "memory");
#endif
#endif
}

__device__ __forceinline__ unsigned short f32_to_bf16(float f) {
  union { float f; unsigned u; } c; c.f = f;
  unsigned u = c.u;
  u += 0x7FFFu + ((u >> 16) & 1u);  // round-to-nearest-even
  return (unsigned short)(u >> 16);
}
__device__ __forceinline__ float bf16_to_f32(unsigned short h) {
  union { unsigned u; float f; } c; c.u = ((unsigned)h) << 16;
  return c.f;
}

// ---------------------------------------------------------------------------
// Elementwise f32 -> bf16 convert
// ---------------------------------------------------------------------------
__global__ void k_f32_to_bf16(const float* __restrict__ in,
                              unsigned short* __restrict__ out, int n) {
  int i = blockIdx.x * blockDim.x + threadIdx.x;
  if (i < n) out[i] = f32_to_bf16(in[i]);
}

// codebook (1024 x 513) f32 -> bf16 with padded row stride 520 (16B-aligned rows)
__global__ void k_cb_to_bf16(const float* __restrict__ cb,
                             unsigned short* __restrict__ out, int n) {
  int i = blockIdx.x * blockDim.x + threadIdx.x;
  if (i < n) {
    int e = i / 513, k = i - e * 513;
    out[(size_t)e * 520 + k] = f32_to_bf16(cb[i]);
  }
}

// ---------------------------------------------------------------------------
// Tiled transpose + convert: in f32 [R][C] -> out bf16 [C][R]
// ---------------------------------------------------------------------------
__global__ __launch_bounds__(256) void k_transpose_bf16(
    const float* __restrict__ in, unsigned short* __restrict__ out,
    int R, int C) {
  __shared__ float tile[32][33];
  int c0 = blockIdx.x * 32, r0 = blockIdx.y * 32;
  for (int j = threadIdx.y; j < 32; j += 8) {
    int r = r0 + j, c = c0 + threadIdx.x;
    if (r < R && c < C) tile[j][threadIdx.x] = in[(size_t)r * C + c];
  }
  __syncthreads();
  for (int j = threadIdx.y; j < 32; j += 8) {
    int c = c0 + j, r = r0 + threadIdx.x;
    if (c < C && r < R) out[(size_t)c * R + r] = f32_to_bf16(tile[threadIdx.x][j]);
  }
}

// ---------------------------------------------------------------------------
// bf16 WMMA GEMM: C[M][N] = act(A[M][K] * Bt[N][K]^T + bias)
// Block tile 128x128, 8 waves, per-wave 32x64 (2x4 WMMA tiles), K step 32.
// Double-buffered LDS; tiles streamed with GLOBAL_LOAD_ASYNC_TO_LDS_B128
// (ASYNCcnt) so tile k+1 loads overlap tile k WMMAs. Steady-state loop is
// branch-free; the last K-tile is peeled.
// ---------------------------------------------------------------------------
template <bool RELU, bool OUT_BF16>
__global__ __launch_bounds__(256) void gemm_bf16_wmma(
    const unsigned short* __restrict__ A, int lda,
    const unsigned short* __restrict__ Bt, int ldb,
    const float* __restrict__ bias, int biasStride,
    void* __restrict__ Cout, int ldc, int K) {
  constexpr int AS = 40;  // padded LDS row stride (bf16 elems)
  __shared__ alignas(16) unsigned short As[2][128 * AS];
  __shared__ alignas(16) unsigned short Bs[2][128 * AS];

  const int t    = threadIdx.x;
  const int lane = t & 31;
  const int wave = t >> 5;
  const int wm   = wave & 3;   // 4 waves along M (32 rows each)
  const int wn   = wave >> 2;  // 2 waves along N (64 cols each)
  const int mBase = blockIdx.x * 128;
  const int nBase = blockIdx.y * 128;

  // Per-thread fill assignment: two 16B chunks per matrix per tile.
  const int r0 = t >> 2;            // rows 0..63 (chunk0) / +64 (chunk1)
  const int kc = (t & 3) * 8;       // K offset within 32-wide tile
  const unsigned short* gA0 = A + (size_t)(mBase + r0) * lda + kc;
  const unsigned short* gA1 = gA0 + (size_t)64 * lda;
  const unsigned short* gB0 = Bt + (size_t)(nBase + r0) * ldb + kc;
  const unsigned short* gB1 = gB0 + (size_t)64 * ldb;
  const int l0 = r0 * AS + kc;
  const int l1 = l0 + 64 * AS;

  v8f acc[2][4];
#pragma unroll
  for (int i = 0; i < 2; i++)
#pragma unroll
    for (int j = 0; j < 4; j++)
#pragma unroll
      for (int e = 0; e < 8; e++) acc[i][j][e] = 0.f;

  // ISA wave32 fragment coordinates
  const int aRow0  = wm * 32 + (lane & 15);
  const int kHalfA = (lane >> 4) * 8;
  const int bRow0  = wn * 64 + (lane & 15);
  const int kHalfB = (lane >> 4) * 8;

  // Issue one 32-wide K tile into buffer `buf`, then advance pointers.
  auto issue = [&](int buf) {
    async_b128(gA0, &As[buf][l0]);
    async_b128(gA1, &As[buf][l1]);
    async_b128(gB0, &Bs[buf][l0]);
    async_b128(gB1, &Bs[buf][l1]);
    gA0 += 32; gA1 += 32; gB0 += 32; gB1 += 32;
  };

  auto compute = [&](int cur) {
    Frag16 a[2], b[4];
#pragma unroll
    for (int mt = 0; mt < 2; mt++) {
      int r = aRow0 + mt * 16;
      a[mt].q[0] = *(const uint4*)&As[cur][r * AS + kHalfA];
      a[mt].q[1] = *(const uint4*)&As[cur][r * AS + 16 + kHalfA];
    }
#pragma unroll
    for (int nt = 0; nt < 4; nt++) {
      int r = bRow0 + nt * 16;
      b[nt].q[0] = *(const uint4*)&Bs[cur][r * AS + kHalfB];
      b[nt].q[1] = *(const uint4*)&Bs[cur][r * AS + 16 + kHalfB];
    }
#pragma unroll
    for (int mt = 0; mt < 2; mt++)
#pragma unroll
      for (int nt = 0; nt < 4; nt++)
        acc[mt][nt] = __builtin_amdgcn_wmma_f32_16x16x32_bf16(
            false, a[mt].v, false, b[nt].v, (short)0, acc[mt][nt], false, false);
  };

  const int nIter = K >> 5;
  issue(0);

  // Steady state: branch-free (prefetch next tile, wait on previous, compute).
#pragma unroll 2
  for (int i = 0; i < nIter - 1; i++) {
    issue((i + 1) & 1);
    wait_async<4>();   // previous tile's 4 async ops complete
    __syncthreads();
    compute(i & 1);
    __syncthreads();   // buffer we just read is refilled next iteration
  }

  // Peeled last tile.
  wait_async<0>();
  __syncthreads();
  compute((nIter - 1) & 1);

  // Epilogue: C/D layout -> per-lane stores with bias/ReLU.
#pragma unroll
  for (int mt = 0; mt < 2; mt++) {
#pragma unroll
    for (int nt = 0; nt < 4; nt++) {
      int gcol = nBase + wn * 64 + nt * 16 + (lane & 15);
      float bv = bias ? bias[(size_t)gcol * biasStride] : 0.f;
#pragma unroll
      for (int r = 0; r < 8; r++) {
        int grow = mBase + wm * 32 + mt * 16 + (lane >> 4) * 8 + r;
        float v = acc[mt][nt][r] + bv;
        if (RELU) v = v > 0.f ? v : 0.f;
        if (OUT_BF16)
          ((unsigned short*)Cout)[(size_t)grow * ldc + gcol] = f32_to_bf16(v);
        else
          ((float*)Cout)[(size_t)grow * ldc + gcol] = v;
      }
    }
  }
}

// ---------------------------------------------------------------------------
// w_hat[:,512] = h2 . W3[:,512] + b3[512]; proj = x . w_hat[:512] + w_hat[512]
// ---------------------------------------------------------------------------
__global__ __launch_bounds__(256) void k_col512_proj(
    const unsigned short* __restrict__ h2b, const float* __restrict__ W3,
    const float* __restrict__ b3, const float* __restrict__ X,
    float* __restrict__ w_hat, float* __restrict__ proj) {
  __shared__ float red[256];
  int row = blockIdx.x, tid = threadIdx.x;

  float s = 0.f;
  for (int k = tid; k < 2048; k += 256)
    s += bf16_to_f32(h2b[(size_t)row * 2048 + k]) * W3[(size_t)k * 513 + 512];
  red[tid] = s;
  __syncthreads();
  for (int st = 128; st > 0; st >>= 1) {
    if (tid < st) red[tid] += red[tid + st];
    __syncthreads();
  }
  float w512 = red[0] + b3[512];
  __syncthreads();
  if (tid == 0) w_hat[(size_t)row * 513 + 512] = w512;

  float s2 = 0.f;
  for (int k = tid; k < 512; k += 256)
    s2 += X[(size_t)row * 512 + k] * w_hat[(size_t)row * 513 + k];
  red[tid] = s2;
  __syncthreads();
  for (int st = 128; st > 0; st >>= 1) {
    if (tid < st) red[tid] += red[tid + st];
    __syncthreads();
  }
  if (tid == 0) proj[row] = red[0] + w512;
}

// ---------------------------------------------------------------------------
// Per-row argmin over 1024 experts of (proj - score)^2, gather codebook row
// into w_tilde (fp32), y_tilde = x . w_tilde[:512] + w_tilde[512].
// ---------------------------------------------------------------------------
__global__ __launch_bounds__(256) void k_argmin_gather(
    const float* __restrict__ scores, const float* __restrict__ proj,
    const float* __restrict__ cb, const float* __restrict__ X,
    float* __restrict__ w_tilde, int* __restrict__ idx_out,
    float* __restrict__ y_out) {
  __shared__ float dmin[256];
  __shared__ int imin[256];
  int row = blockIdx.x, tid = threadIdx.x;
  float p = proj[row];

  float bd = 3.402823466e38f;
  int bi = 0;
  for (int e = tid; e < 1024; e += 256) {
    float d = p - scores[(size_t)row * 1024 + e];
    d = d * d;
    if (d < bd) { bd = d; bi = e; }
  }
  dmin[tid] = bd;
  imin[tid] = bi;
  __syncthreads();
  for (int st = 128; st > 0; st >>= 1) {
    if (tid < st) {
      float od = dmin[tid + st];
      int oi = imin[tid + st];
      if (od < dmin[tid] || (od == dmin[tid] && oi < imin[tid])) {
        dmin[tid] = od;
        imin[tid] = oi;
      }
    }
    __syncthreads();
  }
  int best = imin[0];
  if (tid == 0) idx_out[row] = best;

  float s = 0.f;
  for (int k = tid; k < 513; k += 256) {
    float c = cb[(size_t)best * 513 + k];
    w_tilde[(size_t)row * 513 + k] = c;
    if (k < 512) s += X[(size_t)row * 512 + k] * c;
  }
  __syncthreads();
  dmin[tid] = s;
  __syncthreads();
  for (int st = 128; st > 0; st >>= 1) {
    if (tid < st) dmin[tid] += dmin[tid + st];
    __syncthreads();
  }
  if (tid == 0) y_out[row] = dmin[0] + cb[(size_t)best * 513 + 512];
}

// ---------------------------------------------------------------------------
// Host orchestration
// ---------------------------------------------------------------------------
extern "C" void kernel_launch(void* const* d_in, const int* in_sizes, int n_in,
                              void* d_out, int out_size, void* d_ws,
                              size_t ws_size, hipStream_t stream) {
  const float* X  = (const float*)d_in[0];
  const float* W1 = (const float*)d_in[1];
  const float* b1 = (const float*)d_in[2];
  const float* W2 = (const float*)d_in[3];
  const float* b2 = (const float*)d_in[4];
  const float* W3 = (const float*)d_in[5];
  const float* b3 = (const float*)d_in[6];
  const float* CB = (const float*)d_in[7];

  constexpr int Bsz = 16384, IN = 512, H = 2048, E = 1024, AUG = 513;

  char* ws = (char*)d_ws;
  size_t o = 0;
  auto take = [&](size_t bytes) {
    size_t r = o;
    o += (bytes + 255) & ~(size_t)255;
    return r;
  };
  unsigned short* Xb  = (unsigned short*)(ws + take((size_t)Bsz * IN * 2));
  unsigned short* W1t = (unsigned short*)(ws + take((size_t)H * IN * 2));
  unsigned short* W2t = (unsigned short*)(ws + take((size_t)H * H * 2));
  unsigned short* W3t = (unsigned short*)(ws + take((size_t)AUG * H * 2));
  unsigned short* CBb = (unsigned short*)(ws + take((size_t)E * 520 * 2));
  char* h1mem = ws + take((size_t)Bsz * H * 2);  // 64 MB
  unsigned short* h1b = (unsigned short*)h1mem;
  float* scores = (float*)h1mem;                 // alias: exactly Bsz*E*4 = 64 MB
  unsigned short* h2b = (unsigned short*)(ws + take((size_t)Bsz * H * 2));
  float* proj = (float*)(ws + take((size_t)Bsz * 4));

  float* w_hat   = (float*)d_out;
  float* w_tilde = w_hat + (size_t)Bsz * AUG;
  int*   idx     = (int*)(w_tilde + (size_t)Bsz * AUG);
  float* y       = (float*)(idx + Bsz);

  // 1) Converts / transposes to bf16 operand layouts
  {
    int n = Bsz * IN;
    k_f32_to_bf16<<<(n + 255) / 256, 256, 0, stream>>>(X, Xb, n);
  }
  {
    int n = E * AUG;
    k_cb_to_bf16<<<(n + 255) / 256, 256, 0, stream>>>(CB, CBb, n);
  }
  k_transpose_bf16<<<dim3(H / 32, IN / 32), dim3(32, 8), 0, stream>>>(W1, W1t, IN, H);
  k_transpose_bf16<<<dim3(H / 32, H / 32), dim3(32, 8), 0, stream>>>(W2, W2t, H, H);
  k_transpose_bf16<<<dim3((AUG + 31) / 32, H / 32), dim3(32, 8), 0, stream>>>(W3, W3t, H, AUG);

  // 2) h1 = relu(X @ W1 + b1)          [bf16 out]
  gemm_bf16_wmma<true, true><<<dim3(Bsz / 128, H / 128), 256, 0, stream>>>(
      Xb, IN, W1t, IN, b1, 1, (void*)h1b, H, IN);

  // 3) h2 = relu(h1 @ W2 + b2)         [bf16 out]
  gemm_bf16_wmma<true, true><<<dim3(Bsz / 128, H / 128), 256, 0, stream>>>(
      h1b, H, W2t, H, b2, 1, (void*)h2b, H, H);

  // 4) w_hat[:, :512] = h2 @ W3[:, :512] + b3[:512]   [fp32 out, ldc=513]
  gemm_bf16_wmma<false, false><<<dim3(Bsz / 128, 512 / 128), 256, 0, stream>>>(
      h2b, H, W3t, H, b3, 1, (void*)w_hat, AUG, H);

  // 5) w_hat[:,512] and proj
  k_col512_proj<<<Bsz, 256, 0, stream>>>(h2b, W3, b3, X, w_hat, proj);

  // 6) scores = X @ cb[:, :512]^T + cb[:, 512]  (aug ones-column folds into bias)
  gemm_bf16_wmma<false, false><<<dim3(Bsz / 128, E / 128), 256, 0, stream>>>(
      Xb, IN, CBb, 520, CB + 512, AUG, (void*)scores, E, IN);

  // 7) argmin over experts, gather w_tilde, y_tilde
  k_argmin_gather<<<Bsz, 256, 0, stream>>>(scores, proj, CB, X, w_tilde, idx, y);
}
